// SimpleLSTM_31602369364394
// MI455X (gfx1250) — matrix operand order repaired and checked by reading, not compiled, not run on hardware
//
#include <hip/hip_runtime.h>
#include <hip/hip_bf16.h>

// ---------------------------------------------------------------------------
// LSTM cell on MI455X (gfx1250):
//   g = [x|h0] @ [Wx|Wh]^T + bx        (M=4096, N=4096, K=2048, bf16 WMMA)
//   i,f,o,c~ = gates; c = f*c0 + i*c~; h = o*tanh(c)
// Gate-interleaved weight columns (n' = 4*h + gate) -> fused epilogue.
// K-loop uses CDNA5 async global->LDS copies (ASYNCcnt) with double-buffered
// LDS tiles so DMA overlaps the WMMA stream.
// ---------------------------------------------------------------------------

typedef __attribute__((ext_vector_type(16))) __bf16 v16bf;
typedef __attribute__((ext_vector_type(4)))  __bf16 v4bf;
typedef __attribute__((ext_vector_type(8)))  float  v8f;
typedef __attribute__((ext_vector_type(4)))  int    v4i;

#define Bsz   4096
#define INsz  1024
#define Hsz   1024
#define Ksz   2048   // IN + H
#define Nsz   4096   // 4*H, gate-interleaved
#define HALF_OUT (Bsz * Hsz)

#if __has_builtin(__builtin_amdgcn_global_load_async_to_lds_b128)
#define HAVE_ASYNC_LDS 1
#else
#define HAVE_ASYNC_LDS 0
#endif

typedef __attribute__((address_space(1))) v4i as1_v4i;  // global int4
typedef __attribute__((address_space(3))) v4i as3_v4i;  // LDS int4

__device__ __forceinline__ void copy16_g2l(const void* g, void* l) {
#if HAVE_ASYNC_LDS
  __builtin_amdgcn_global_load_async_to_lds_b128((as1_v4i*)g, (as3_v4i*)l, 0, 0);
#else
  *(uint4*)l = *(const uint4*)g;
#endif
}

__device__ __forceinline__ void wait_g2l() {
#if HAVE_ASYNC_LDS
  asm volatile("s_wait_asynccnt 0x0" ::: "memory");
#endif
}

// -------------------------- pack A: [x|h0] -> bf16 --------------------------
__global__ __launch_bounds__(256) void lstm_pack_a(const float* __restrict__ x,
                                                   const float* __restrict__ h0,
                                                   __bf16* __restrict__ A) {
  int id = blockIdx.x * 256 + threadIdx.x;          // 4096*512 quads
  int b  = id >> 9;
  int k  = (id & 511) << 2;
  float4 v;
  if (k < INsz) v = *(const float4*)(x  + (size_t)b * INsz + k);
  else          v = *(const float4*)(h0 + (size_t)b * Hsz + (k - INsz));
  v4bf o;
  o[0] = (__bf16)v.x; o[1] = (__bf16)v.y; o[2] = (__bf16)v.z; o[3] = (__bf16)v.w;
  *(v4bf*)(A + (size_t)b * Ksz + k) = o;
}

// ---------------- pack B: gate-interleaved [4096][2048] bf16 ----------------
__global__ __launch_bounds__(256) void lstm_pack_b(
    const float* __restrict__ Wxi, const float* __restrict__ Wxf,
    const float* __restrict__ Wxo, const float* __restrict__ Wxc,
    const float* __restrict__ Whi, const float* __restrict__ Whf,
    const float* __restrict__ Who, const float* __restrict__ Whc,
    __bf16* __restrict__ Bp) {
  int id   = blockIdx.x * 256 + threadIdx.x;        // 4096*512 quads
  int n    = id >> 9;                               // n' = 4*h + gate
  int k    = (id & 511) << 2;
  int gate = n & 3;
  int h    = n >> 2;
  const float* W;
  int off;
  if (k < INsz) {
    W = (gate == 0) ? Wxi : (gate == 1) ? Wxf : (gate == 2) ? Wxo : Wxc;
    off = h * INsz + k;
  } else {
    W = (gate == 0) ? Whi : (gate == 1) ? Whf : (gate == 2) ? Who : Whc;
    off = h * Hsz + (k - INsz);
  }
  float4 v = *(const float4*)(W + off);
  v4bf o;
  o[0] = (__bf16)v.x; o[1] = (__bf16)v.y; o[2] = (__bf16)v.z; o[3] = (__bf16)v.w;
  *(v4bf*)(Bp + (size_t)n * Ksz + k) = o;
}

// ----------------------------- fused GEMM+cell ------------------------------
// Block tile: 128(M) x 128(N'), 8 waves, wave tile 64x32 (4x2 WMMA tiles).
// LDS: double-buffered A/B tiles (4 x 8KB); reused as 128x132 f32 staging
// buffer (66KB) for the fused epilogue.
__device__ __forceinline__ float sigmoid_f(float v) {
  return 1.0f / (1.0f + __expf(-v));
}

__global__ __launch_bounds__(256) void lstm_gemm_cell(
    const __bf16* __restrict__ A, const __bf16* __restrict__ Bp,
    const float* __restrict__ bxi, const float* __restrict__ bxf,
    const float* __restrict__ bxo, const float* __restrict__ bxc,
    const float* __restrict__ c0, float* __restrict__ out) {
  __shared__ __align__(16) unsigned char smem[128 * 132 * 4];  // 67584 B
  // buffers: A0 @0, A1 @8192, B0 @16384, B1 @24576 ; gS overlays everything
  float* gS = (float*)smem;

  const int tid  = threadIdx.x;
  const int lane = tid & 31;
  const int w    = tid >> 5;          // 8 waves
  const int l    = lane & 15;
  const int half = lane >> 4;
  const int waveM = (w & 1) * 64;     // 2 waves along M
  const int waveN = (w >> 1) * 32;    // 4 waves along N
  const int M0 = blockIdx.y * 128;
  const int N0 = blockIdx.x * 128;

  v8f acc[4][2];
#pragma unroll
  for (int mt = 0; mt < 4; ++mt)
#pragma unroll
    for (int nt = 0; nt < 2; ++nt) acc[mt][nt] = (v8f)0.0f;

  // tile copy: 128 rows x 64B per tile = 512 x 16B chunks; 2 chunks/thread
  const int r0 = tid >> 2,         s0 = tid & 3;
  const int r1 = (tid + 256) >> 2, s1 = (tid + 256) & 3;
  const __bf16* gA0 = A  + (size_t)(M0 + r0) * Ksz + s0 * 8;
  const __bf16* gA1 = A  + (size_t)(M0 + r1) * Ksz + s1 * 8;
  const __bf16* gB0 = Bp + (size_t)(N0 + r0) * Ksz + s0 * 8;
  const __bf16* gB1 = Bp + (size_t)(N0 + r1) * Ksz + s1 * 8;
  const int lA0 = r0 * 64 + s0 * 16;
  const int lA1 = r1 * 64 + s1 * 16;
  const int lB0 = 16384 + r0 * 64 + s0 * 16;
  const int lB1 = 16384 + r1 * 64 + s1 * 16;

  // prologue: async-load k-tile 0 into buffer 0
  copy16_g2l(gA0, smem + lA0);
  copy16_g2l(gA1, smem + lA1);
  copy16_g2l(gB0, smem + lB0);
  copy16_g2l(gB1, smem + lB1);
  wait_g2l();
  __syncthreads();

  for (int kt = 0; kt < Ksz / 32; ++kt) {
    const int cur = kt & 1;
    // issue async loads for tile kt+1 into the other buffer (overlaps WMMA)
    if (kt + 1 < Ksz / 32) {
      const int k   = (kt + 1) * 32;
      const int nxt = (cur ^ 1) * 8192;
      copy16_g2l(gA0 + k, smem + nxt + lA0);
      copy16_g2l(gA1 + k, smem + nxt + lA1);
      copy16_g2l(gB0 + k, smem + nxt + lB0);
      copy16_g2l(gB1 + k, smem + nxt + lB1);
    }

    const unsigned char* AsU = smem + cur * 8192;
    const unsigned char* BsU = smem + 16384 + cur * 8192;

    // A fragments: ISA 16-bit A 16x32 layout — lane<16: K0..7 & K16..23,
    // lane>=16: K8..15 & K24..31 (two ds_load_b128 each)
    v16bf af[4];
#pragma unroll
    for (int mt = 0; mt < 4; ++mt) {
      int base = (waveM + mt * 16 + l) * 64 + half * 16;
      uint4* p = (uint4*)&af[mt];
      p[0] = *(const uint4*)(AsU + base);
      p[1] = *(const uint4*)(AsU + base + 32);
    }
    // B fragments: lane<16: col=l K0..7 & K16..23; lane>=16: col=l K8..15 & K24..31
    v16bf bfr[2];
#pragma unroll
    for (int nt = 0; nt < 2; ++nt) {
      int base = (waveN + nt * 16 + l) * 64 + half * 16;
      uint4* p = (uint4*)&bfr[nt];
      p[0] = *(const uint4*)(BsU + base);
      p[1] = *(const uint4*)(BsU + base + 32);
    }
#pragma unroll
    for (int mt = 0; mt < 4; ++mt)
#pragma unroll
      for (int nt = 0; nt < 2; ++nt)
        acc[mt][nt] = __builtin_amdgcn_wmma_f32_16x16x32_bf16(
            false, af[mt], false, bfr[nt], (short)0, acc[mt][nt], false, false);

    wait_g2l();       // next tile landed in LDS
    __syncthreads();  // all waves done reading cur / writing nxt
  }

  // ---- fused epilogue: accumulators -> LDS -> gate math -> h_t, c_t ----
#pragma unroll
  for (int mt = 0; mt < 4; ++mt)
#pragma unroll
    for (int nt = 0; nt < 2; ++nt) {
      int Nc = waveN + nt * 16 + l;
#pragma unroll
      for (int j = 0; j < 8; ++j) {
        int Mr = waveM + mt * 16 + j + half * 8;  // C/D layout: VGPR j, half
        gS[Mr * 132 + Nc] = acc[mt][nt][j];
      }
    }
  __syncthreads();

  const int hl = tid & 31;                 // h within tile [0,32)
  const int hg = blockIdx.x * 32 + hl;     // global h
#pragma unroll 4
  for (int r = 0; r < 16; ++r) {
    int m  = (tid >> 5) + r * 8;
    int mg = M0 + m;
    float4 q = *(const float4*)(gS + m * 132 + hl * 4);  // gates i,f,o,c
    float gi = q.x + bxi[hg];
    float gf = q.y + bxf[hg];
    float go = q.z + bxo[hg];
    float gc = q.w + bxc[hg];
    float it = sigmoid_f(gi);
    float ft = sigmoid_f(gf);
    float ot = sigmoid_f(go);
    float ct = ft * c0[(size_t)mg * Hsz + hg] + it * tanhf(gc);
    float ht = ot * tanhf(ct);
    out[(size_t)mg * Hsz + hg]            = ht;
    out[HALF_OUT + (size_t)mg * Hsz + hg] = ct;
  }
}

// ------------------------------- launcher -----------------------------------
extern "C" void kernel_launch(void* const* d_in, const int* in_sizes, int n_in,
                              void* d_out, int out_size, void* d_ws, size_t ws_size,
                              hipStream_t stream) {
  // setup_inputs() order:
  // 0 x, 1 h0, 2 c0, 3 W_xi, 4 b_xi, 5 W_xf, 6 b_xf, 7 W_xo, 8 b_xo,
  // 9 W_xc, 10 b_xc, 11 W_hi, 12 W_hf, 13 W_ho, 14 W_hc
  const float* x   = (const float*)d_in[0];
  const float* h0  = (const float*)d_in[1];
  const float* c0  = (const float*)d_in[2];
  const float* Wxi = (const float*)d_in[3];
  const float* bxi = (const float*)d_in[4];
  const float* Wxf = (const float*)d_in[5];
  const float* bxf = (const float*)d_in[6];
  const float* Wxo = (const float*)d_in[7];
  const float* bxo = (const float*)d_in[8];
  const float* Wxc = (const float*)d_in[9];
  const float* bxc = (const float*)d_in[10];
  const float* Whi = (const float*)d_in[11];
  const float* Whf = (const float*)d_in[12];
  const float* Who = (const float*)d_in[13];
  const float* Whc = (const float*)d_in[14];
  float* out = (float*)d_out;

  char* ws = (char*)d_ws;
  __bf16* A  = (__bf16*)ws;                              // 16 MiB
  __bf16* Bp = (__bf16*)(ws + (size_t)Bsz * Ksz * 2);    // 16 MiB

  lstm_pack_a<<<(Bsz * (Ksz / 4)) / 256, 256, 0, stream>>>(x, h0, A);
  lstm_pack_b<<<(Nsz * (Ksz / 4)) / 256, 256, 0, stream>>>(
      Wxi, Wxf, Wxo, Wxc, Whi, Whf, Who, Whc, Bp);
  lstm_gemm_cell<<<dim3(Nsz / 128, Bsz / 128), 256, 0, stream>>>(
      A, Bp, bxi, bxf, bxo, bxc, c0, out);
}
